// SelfAttention_66769561584229
// MI455X (gfx1250) — matrix operand order repaired and checked by reading, not compiled, not run on hardware
//
#include <hip/hip_runtime.h>

// ---------------------------------------------------------------------------
// MI455X (gfx1250) fused GroupNorm + single-head self-attention, bf16 WMMA.
// groupnorm -> hT | weight precvt -> bf16 | qkv GEMM (2x2 blocked) ->
// qT(scaled),kT,v | flash-attn -> aoT | out-proj GEMM + bias + residual.
// ---------------------------------------------------------------------------

typedef __attribute__((ext_vector_type(16))) __bf16 v16bf;
typedef __attribute__((ext_vector_type(8)))  __bf16 v8bf;
typedef __attribute__((ext_vector_type(8)))  float  v8f;

#define B_    8
#define C_    512
#define N_    4096
#define CPG   16
#define EPSV  1e-5f
#define QK_SCALE 0.04419417382415922f   // 1/sqrt(512)

static __device__ __forceinline__ v8f wmma_bf16(v16bf a, v16bf b, v8f c) {
    return __builtin_amdgcn_wmma_f32_16x16x32_bf16(false, a, false, b,
                                                   (short)0, c, false, false);
}

// --- A-fragment (16x32 bf16) from row-major bf16 [row][k], k contiguous -----
static __device__ __forceinline__ v16bf load_a_bf16(const __bf16* src, int ld, int lane) {
    const int half = lane >> 4, m = lane & 15;
    const __bf16* p = src + (size_t)m * ld;
    union { v16bf v; v8bf h[2]; } u;
    u.h[0] = *(const v8bf*)(p + (half ? 16 : 0));
    u.h[1] = *(const v8bf*)(p + (half ? 24 : 8));
    return u.v;
}

// --- B-fragment (32x16 bf16) from "n-major" bf16 src[n][k], k contiguous ----
static __device__ __forceinline__ v16bf load_b_bf16(const __bf16* src, int ld, int lane) {
    const int half = lane >> 4, n = lane & 15;
    const __bf16* p = src + (size_t)n * ld + half * 16;
    union { v16bf v; v8bf h[2]; } u;
    u.h[0] = *(const v8bf*)(p);
    u.h[1] = *(const v8bf*)(p + 8);
    return u.v;
}

// ---------------------------------------------------------------------------
// Kernel 0: one-shot f32 -> bf16 weight conversion (removes per-tile cvt).
// ---------------------------------------------------------------------------
__global__ __launch_bounds__(256)
void wconv_kernel(const float* __restrict__ wq, const float* __restrict__ wo,
                  __bf16* __restrict__ wqb, __bf16* __restrict__ wob) {
    const int i = blockIdx.x * 256 + threadIdx.x;
    const int NQ = 3 * C_ * C_;
    if (i < NQ)            wqb[i] = (__bf16)wq[i];
    else                   wob[i - NQ] = (__bf16)wo[i - NQ];
}

// ---------------------------------------------------------------------------
// Kernel 1: GroupNorm. One block per (batch, group). Writes hT[b][n][c] bf16.
// ---------------------------------------------------------------------------
__global__ __launch_bounds__(256)
void gn_kernel(const float* __restrict__ x, const float* __restrict__ gamma,
               const float* __restrict__ beta, __bf16* __restrict__ hT) {
    const int b = blockIdx.x >> 5, g = blockIdx.x & 31;
    const int tid = threadIdx.x;
    const float* xs = x + ((size_t)b * C_ + g * CPG) * N_;

    float s = 0.f, ss = 0.f;
    for (int i = tid; i < CPG * N_; i += 256) {
        const float v = xs[i];
        s += v; ss += v * v;
    }
    __shared__ float rs[256], rss[256];
    rs[tid] = s; rss[tid] = ss;
    __syncthreads();
    for (int off = 128; off; off >>= 1) {
        if (tid < off) { rs[tid] += rs[tid + off]; rss[tid] += rss[tid + off]; }
        __syncthreads();
    }
    const float inv_cnt = 1.f / (float)(CPG * N_);
    const float mean = rs[0] * inv_cnt;
    const float var  = rss[0] * inv_cnt - mean * mean;
    const float rstd = rsqrtf(var + EPSV);

    float gsc[CPG], gbi[CPG];
#pragma unroll
    for (int c = 0; c < CPG; ++c) {
        const float gm = gamma[g * CPG + c];
        gsc[c] = gm * rstd;
        gbi[c] = beta[g * CPG + c] - mean * gsc[c];
    }
    __bf16* hb = hT + (size_t)b * N_ * C_ + g * CPG;
    for (int n = tid; n < N_; n += 256) {
        union { v16bf v; __bf16 e[16]; } o;
#pragma unroll
        for (int c = 0; c < CPG; ++c)
            o.e[c] = (__bf16)(xs[(size_t)c * N_ + n] * gsc[c] + gbi[c]);
        *(v16bf*)(hb + (size_t)n * C_) = o.v;
    }
}

// ---------------------------------------------------------------------------
// Kernel 2: QKV GEMM, 2x2 register blocked. Wave tile 32(o) x 32(n);
// block = 8 waves (2 x 4) => 64(o) x 128(n). 4 WMMAs per 4 fragment loads.
// ---------------------------------------------------------------------------
__global__ __launch_bounds__(256)
void qkv_kernel(const __bf16* __restrict__ w, const float* __restrict__ bias,
                const __bf16* __restrict__ hT,
                __bf16* __restrict__ qT, __bf16* __restrict__ kT,
                __bf16* __restrict__ vv) {
    const int b = blockIdx.z;
    const int wave = threadIdx.x >> 5, lane = threadIdx.x & 31;
    const int o0 = blockIdx.y * 64 + (wave >> 2) * 32;
    const int n0 = blockIdx.x * 128 + (wave & 3) * 32;
    const __bf16* hb = hT + (size_t)b * N_ * C_;

    v8f acc00 = {0.f,0.f,0.f,0.f,0.f,0.f,0.f,0.f};
    v8f acc01 = acc00, acc10 = acc00, acc11 = acc00;
#pragma unroll 4
    for (int cw = 0; cw < C_ / 32; ++cw) {
        v16bf a0 = load_a_bf16(w + (size_t)o0 * C_ + cw * 32, C_, lane);
        v16bf a1 = load_a_bf16(w + (size_t)(o0 + 16) * C_ + cw * 32, C_, lane);
        v16bf b0 = load_b_bf16(hb + (size_t)n0 * C_ + cw * 32, C_, lane);
        v16bf b1 = load_b_bf16(hb + (size_t)(n0 + 16) * C_ + cw * 32, C_, lane);
        acc00 = wmma_bf16(a0, b0, acc00);
        acc01 = wmma_bf16(a0, b1, acc01);
        acc10 = wmma_bf16(a1, b0, acc10);
        acc11 = wmma_bf16(a1, b1, acc11);
    }

    const int half = lane >> 4, lid = lane & 15;
    v8f accs[2][2] = {{acc00, acc01}, {acc10, acc11}};
#pragma unroll
    for (int oi = 0; oi < 2; ++oi) {
#pragma unroll
        for (int ni = 0; ni < 2; ++ni) {
            const int ob = o0 + oi * 16 + half * 8;
            const int n  = n0 + ni * 16 + lid;
            float vals[8];
#pragma unroll
            for (int r = 0; r < 8; ++r) vals[r] = accs[oi][ni][r] + bias[ob + r];
            if (o0 < C_) {                       // Q -> qT[b][n][c], pre-scaled
                union { v8bf v; __bf16 e[8]; } st;
#pragma unroll
                for (int r = 0; r < 8; ++r) st.e[r] = (__bf16)(vals[r] * QK_SCALE);
                *(v8bf*)(qT + ((size_t)b * N_ + n) * C_ + ob) = st.v;
            } else if (o0 < 2 * C_) {            // K -> kT[b][n][c]
                union { v8bf v; __bf16 e[8]; } st;
#pragma unroll
                for (int r = 0; r < 8; ++r) st.e[r] = (__bf16)vals[r];
                *(v8bf*)(kT + ((size_t)b * N_ + n) * C_ + (ob - C_)) = st.v;
            } else {                             // V -> v[b][c][n]
#pragma unroll
                for (int r = 0; r < 8; ++r)
                    vv[((size_t)b * C_ + (ob - 2 * C_) + r) * N_ + n] = (__bf16)vals[r];
            }
        }
    }
}

// ---------------------------------------------------------------------------
// Kernel 3: flash attention. Block: 8 waves = 2 query-tiles x 4 c-slices(128).
// ---------------------------------------------------------------------------
__global__ __launch_bounds__(256)
void attn_kernel(const __bf16* __restrict__ qT, const __bf16* __restrict__ kT,
                 const __bf16* __restrict__ vv, __bf16* __restrict__ aoT) {
    const int b = blockIdx.y;
    const int wave = threadIdx.x >> 5, lane = threadIdx.x & 31;
    const int qt = wave >> 2, cs = wave & 3;
    const int q0 = blockIdx.x * 32 + qt * 16;
    const int cslice = cs * 128;
    const int half = lane >> 4, lid = lane & 15;

    const __bf16* qb = qT + (size_t)b * N_ * C_;
    const __bf16* kb = kT + (size_t)b * N_ * C_;
    const __bf16* vb = vv + (size_t)b * C_ * N_;

    __shared__ float  sred0[8][256];
    __shared__ float  sred1[8][256];
    __shared__ __bf16 sp[2][16 * 32];

    v16bf qf[4];
#pragma unroll
    for (int cw = 0; cw < 4; ++cw)
        qf[cw] = load_a_bf16(qb + (size_t)q0 * C_ + cslice + cw * 32, C_, lane);

    v8f O[8];
#pragma unroll
    for (int t = 0; t < 8; ++t) O[t] = (v8f){0.f,0.f,0.f,0.f,0.f,0.f,0.f,0.f};
    float mrow[8], lrow[8];
#pragma unroll
    for (int r = 0; r < 8; ++r) { mrow[r] = -1e30f; lrow[r] = 0.f; }

    for (int jb = 0; jb < N_; jb += 32) {
        v8f s0 = {0.f,0.f,0.f,0.f,0.f,0.f,0.f,0.f};
        v8f s1 = {0.f,0.f,0.f,0.f,0.f,0.f,0.f,0.f};
#pragma unroll
        for (int cw = 0; cw < 4; ++cw) {
            v16bf k0 = load_b_bf16(kb + (size_t)jb * C_ + cslice + cw * 32, C_, lane);
            v16bf k1 = load_b_bf16(kb + (size_t)(jb + 16) * C_ + cslice + cw * 32, C_, lane);
            s0 = wmma_bf16(qf[cw], k0, s0);
            s1 = wmma_bf16(qf[cw], k1, s1);
        }
        if (jb + 32 < N_)
            __builtin_prefetch(kb + (size_t)(jb + 32) * C_ + cslice, 0, 0);

#pragma unroll
        for (int r = 0; r < 8; ++r) {
            sred0[wave][r * 32 + lane] = s0[r];
            sred1[wave][r * 32 + lane] = s1[r];
        }
        __syncthreads();
        float S0[8], S1[8];
#pragma unroll
        for (int r = 0; r < 8; ++r) {
            const int idx = r * 32 + lane, wq = qt * 4;
            S0[r] = sred0[wq][idx] + sred0[wq+1][idx] + sred0[wq+2][idx] + sred0[wq+3][idx];
            S1[r] = sred1[wq][idx] + sred1[wq+1][idx] + sred1[wq+2][idx] + sred1[wq+3][idx];
        }
        __syncthreads();

        float P0[8], P1[8], scal[8];
#pragma unroll
        for (int r = 0; r < 8; ++r) {
            float rm = fmaxf(S0[r], S1[r]);
            rm = fmaxf(rm, __shfl_xor(rm, 1, 16));
            rm = fmaxf(rm, __shfl_xor(rm, 2, 16));
            rm = fmaxf(rm, __shfl_xor(rm, 4, 16));
            rm = fmaxf(rm, __shfl_xor(rm, 8, 16));
            const float mn = fmaxf(mrow[r], rm);
            scal[r] = __expf(mrow[r] - mn);
            P0[r] = __expf(S0[r] - mn);
            P1[r] = __expf(S1[r] - mn);
            float rsum = P0[r] + P1[r];
            rsum += __shfl_xor(rsum, 1, 16);
            rsum += __shfl_xor(rsum, 2, 16);
            rsum += __shfl_xor(rsum, 4, 16);
            rsum += __shfl_xor(rsum, 8, 16);
            lrow[r] = lrow[r] * scal[r] + rsum;
            mrow[r] = mn;
        }
#pragma unroll
        for (int t = 0; t < 8; ++t)
#pragma unroll
            for (int r = 0; r < 8; ++r) O[t][r] *= scal[r];

        if (cs == 0) {
#pragma unroll
            for (int r = 0; r < 8; ++r) {
                const int row = r + 8 * half;
                sp[qt][row * 32 + lid]      = (__bf16)P0[r];
                sp[qt][row * 32 + 16 + lid] = (__bf16)P1[r];
            }
        }
        __syncthreads();

        v16bf pf = load_a_bf16(&sp[qt][0], 32, lane);
#pragma unroll
        for (int t = 0; t < 8; ++t) {
            v16bf vf = load_b_bf16(vb + (size_t)(cslice + t * 16) * N_ + jb, N_, lane);
            O[t] = wmma_bf16(pf, vf, O[t]);
        }
    }

    float inv[8];
#pragma unroll
    for (int r = 0; r < 8; ++r) inv[r] = 1.f / lrow[r];
    __bf16* ao = aoT + (size_t)b * N_ * C_;
#pragma unroll
    for (int t = 0; t < 8; ++t) {
        const int c = cslice + t * 16 + lid;
#pragma unroll
        for (int r = 0; r < 8; ++r) {
            const int row = q0 + r + 8 * half;
            ao[(size_t)row * C_ + c] = (__bf16)(O[t][r] * inv[r]);
        }
    }
}

// ---------------------------------------------------------------------------
// Kernel 4: out projection + bias + residual, 2x2 register blocked.
// ---------------------------------------------------------------------------
__global__ __launch_bounds__(256)
void proj_kernel(const __bf16* __restrict__ w, const float* __restrict__ bias,
                 const __bf16* __restrict__ aoT, const float* __restrict__ x,
                 float* __restrict__ out) {
    const int b = blockIdx.z;
    const int wave = threadIdx.x >> 5, lane = threadIdx.x & 31;
    const int o0 = blockIdx.y * 64 + (wave >> 2) * 32;
    const int n0 = blockIdx.x * 128 + (wave & 3) * 32;
    const __bf16* ab = aoT + (size_t)b * N_ * C_;

    v8f acc00 = {0.f,0.f,0.f,0.f,0.f,0.f,0.f,0.f};
    v8f acc01 = acc00, acc10 = acc00, acc11 = acc00;
#pragma unroll 4
    for (int cw = 0; cw < C_ / 32; ++cw) {
        v16bf a0 = load_a_bf16(w + (size_t)o0 * C_ + cw * 32, C_, lane);
        v16bf a1 = load_a_bf16(w + (size_t)(o0 + 16) * C_ + cw * 32, C_, lane);
        v16bf b0 = load_b_bf16(ab + (size_t)n0 * C_ + cw * 32, C_, lane);
        v16bf b1 = load_b_bf16(ab + (size_t)(n0 + 16) * C_ + cw * 32, C_, lane);
        acc00 = wmma_bf16(a0, b0, acc00);
        acc01 = wmma_bf16(a0, b1, acc01);
        acc10 = wmma_bf16(a1, b0, acc10);
        acc11 = wmma_bf16(a1, b1, acc11);
    }
    const int half = lane >> 4, lid = lane & 15;
    v8f accs[2][2] = {{acc00, acc01}, {acc10, acc11}};
#pragma unroll
    for (int oi = 0; oi < 2; ++oi) {
#pragma unroll
        for (int ni = 0; ni < 2; ++ni) {
            const int n = n0 + ni * 16 + lid;
#pragma unroll
            for (int r = 0; r < 8; ++r) {
                const int o = o0 + oi * 16 + half * 8 + r;
                const size_t idx = ((size_t)b * C_ + o) * N_ + n;
                out[idx] = x[idx] + accs[oi][ni][r] + bias[o];
            }
        }
    }
}

// ---------------------------------------------------------------------------
extern "C" void kernel_launch(void* const* d_in, const int* in_sizes, int n_in,
                              void* d_out, int out_size, void* d_ws, size_t ws_size,
                              hipStream_t stream) {
    const float* x     = (const float*)d_in[0];
    const float* gamma = (const float*)d_in[1];
    const float* beta  = (const float*)d_in[2];
    const float* w_qkv = (const float*)d_in[3];
    const float* b_qkv = (const float*)d_in[4];
    const float* w_out = (const float*)d_in[5];
    const float* b_out = (const float*)d_in[6];

    const size_t TSZ = (size_t)B_ * N_ * C_;   // 16.7M bf16 elems = 32 MB
    __bf16* hT   = (__bf16*)d_ws;
    __bf16* qT   = hT  + TSZ;
    __bf16* kT   = qT  + TSZ;
    __bf16* vv   = kT  + TSZ;
    __bf16* aoT  = vv  + TSZ;
    __bf16* wq_b = aoT + TSZ;                  // 1.5 MB
    __bf16* wo_b = wq_b + (size_t)3 * C_ * C_; // 0.5 MB

    wconv_kernel<<<dim3((4 * C_ * C_) / 256), 256, 0, stream>>>(w_qkv, w_out, wq_b, wo_b);
    gn_kernel   <<<dim3(B_ * 32), 256, 0, stream>>>(x, gamma, beta, hT);
    qkv_kernel  <<<dim3(N_/128, (3*C_)/64, B_), 256, 0, stream>>>(wq_b, b_qkv, hT, qT, kT, vv);
    attn_kernel <<<dim3(N_/32, B_), 256, 0, stream>>>(qT, kT, vv, aoT);
    proj_kernel <<<dim3(N_/128, C_/64, B_), 256, 0, stream>>>(wo_b, b_out, aoT, x, (float*)d_out);
}
